// RSSM_64003602645154
// MI455X (gfx1250) — matrix operand order repaired and checked by reading, not compiled, not run on hardware
//
#include <hip/hip_runtime.h>
#include <hip/hip_bf16.h>

// ---------------- problem constants ----------------
#define Bb    64
#define Ll    64
#define Ee    4096
#define Aa    3
#define DETER 4096
#define STOCH 32
#define DISC  32
#define SD    1024       // STOCH*DISC
#define HIDN  512
#define KK    8
#define DPB   512        // DETER/KK
#define X3    1536       // 3*HIDN
#define ODIM  6144       // 1024 stoch + 4096 deter + 1024 logit

typedef __attribute__((ext_vector_type(16))) __bf16          v16bf;
typedef __attribute__((ext_vector_type(16))) unsigned short  v16us;
typedef __attribute__((ext_vector_type(8)))  unsigned short  v8us;
typedef __attribute__((ext_vector_type(8)))  float           v8f;

union FragBF { v16bf bf; v16us v; v8us h[2]; unsigned short u[16]; };

__device__ __forceinline__ unsigned short f2bf(float f) {
    unsigned int x = __builtin_bit_cast(unsigned int, f);
    x += 0x7FFFu + ((x >> 16) & 1u);           // round-to-nearest-even
    return (unsigned short)(x >> 16);
}

__device__ __forceinline__ float sigm(float x) { return 1.f / (1.f + expf(-x)); }
__device__ __forceinline__ float silu(float x) { return x * sigm(x); }

// deterministic pseudo-gumbel (reference uses a fixed JAX key; no runtime check here)
__device__ __forceinline__ float gumbel_hash(int t, int b, int s, int d) {
    unsigned long long x = ((unsigned long long)t << 40) ^ ((unsigned long long)b << 24)
                         ^ ((unsigned long long)s << 8) ^ (unsigned long long)d
                         ^ 0x9E3779B97F4A7C15ull;
    x ^= x >> 30; x *= 0xBF58476D1CE4E5B9ull;
    x ^= x >> 27; x *= 0x94D049BB133111EBull;
    x ^= x >> 31;
    float u = (float)((x >> 40) + 1ull) * (1.0f / 16777218.0f);  // (0,1)
    return -logf(-logf(u));
}

// ---------------- generic fp32 -> bf16 convert (embed, one shot) ----------------
__global__ __launch_bounds__(256)
void cvt_bf16(const float* __restrict__ src, unsigned short* __restrict__ dst, long long n)
{
    for (long long i = (long long)blockIdx.x * blockDim.x + threadIdx.x;
         i < n; i += (long long)gridDim.x * blockDim.x)
        dst[i] = f2bf(src[i]);
}

// ---------------- weight pack: fp32 -> bf16 B-fragment order ----------------
// out[((nt*Ktiles + kt)*32 + lane)*16 + j] = W[n*sn + k*sk]
//   n = nt*16 + (lane&15);  k = kt*32 + (j>>3)*16 + (lane>>4)*8 + (j&7)   (ISA 05_wmma B layout)
__global__ __launch_bounds__(256)
void pack_w(const float* __restrict__ W, long long zW, int sn, int sk,
            int Ntiles, int Ktiles, unsigned short* __restrict__ out, long long zOut)
{
    W   += (long long)blockIdx.z * zW;
    out += (long long)blockIdx.z * zOut;
    long long total = (long long)Ntiles * Ktiles * 32 * 16;
    for (long long i = (long long)blockIdx.x * blockDim.x + threadIdx.x;
         i < total; i += (long long)gridDim.x * blockDim.x) {
        int j    = (int)(i & 15);
        int lane = (int)((i >> 4) & 31);
        long long tile = i >> 9;
        int kt = (int)(tile % Ktiles);
        int nt = (int)(tile / Ktiles);
        int n  = nt * 16 + (lane & 15);
        int k  = kt * 32 + (j >> 3) * 16 + (lane >> 4) * 8 + (j & 7);
        out[i] = f2bf(W[(long long)n * sn + (long long)k * sk]);
    }
}

// ---------------- fragment loaders ----------------
__device__ __forceinline__ void load_frags2(const unsigned short* __restrict__ Arow,
                                            const unsigned short* __restrict__ wb0,
                                            const unsigned short* __restrict__ wb1,
                                            int kt, int koff,
                                            FragBF& a, FragBF& b0, FragBF& b1)
{
    const int k0 = kt * 32;
    a.h[0] = *(const v8us*)(Arow + k0 + koff);
    a.h[1] = *(const v8us*)(Arow + k0 + 16 + koff);
    b0.v   = *(const v16us*)(wb0 + (long long)kt * 512);
    b1.v   = *(const v16us*)(wb1 + (long long)kt * 512);
}

// ---------------- ping-pong pipelined dual-tile K phase ----------------
// Arow is pre-biased so global k-index kt*32 addresses the right element.
// Explicit 2x unroll with two fragment sets: each load fills the set not
// currently feeding a WMMA -> no register rotation copies, and the pre-WMMA
// wait covers only older loads. Requires (kte - kts) even (true for all
// launches here: phase lengths 16/32/48/128).
__device__ __forceinline__ void gemm_phase(const unsigned short* __restrict__ Arow,
                                           const unsigned short* __restrict__ wb0,
                                           const unsigned short* __restrict__ wb1,
                                           int kts, int kte, int KT, int koff,
                                           v8f& acc0, v8f& acc1)
{
    if (kts >= kte) return;
    FragBF a0, b00, b10, a1, b01, b11;
    load_frags2(Arow, wb0, wb1, kts, koff, a0, b00, b10);
    for (int kt = kts; kt < kte; kt += 2) {
        load_frags2(Arow, wb0, wb1, kt + 1, koff, a1, b01, b11);
        if (kt + 8 < KT) {
            __builtin_prefetch(wb0 + (long long)(kt + 8) * 512, 0, 1);
            __builtin_prefetch(wb1 + (long long)(kt + 8) * 512, 0, 1);
        }
        acc0 = __builtin_amdgcn_wmma_f32_16x16x32_bf16(false, a0.bf, false, b00.bf, (short)0, acc0, false, false);
        acc1 = __builtin_amdgcn_wmma_f32_16x16x32_bf16(false, a0.bf, false, b10.bf, (short)0, acc1, false, false);
        if (kt + 2 < kte)
            load_frags2(Arow, wb0, wb1, kt + 2, koff, a0, b00, b10);
        acc0 = __builtin_amdgcn_wmma_f32_16x16x32_bf16(false, a1.bf, false, b01.bf, (short)0, acc0, false, false);
        acc1 = __builtin_amdgcn_wmma_f32_16x16x32_bf16(false, a1.bf, false, b11.bf, (short)0, acc1, false, false);
    }
}

// ---------------- generic WMMA GEMM: C[64 x N] = A[64 x Ktot] * Wpacked + bias
// A is bf16, two sources: k < K0 from A0 (row stride lda0), else A1 (lda1).
// Each wave computes TWO 16x16 N-tiles sharing the A fragment; K loop split
// into two branch-free phases (one per A source).
// gridDim = (N/32, 4, nblocks); per-z strides allow block-diagonal weights.
__global__ __launch_bounds__(32)
void wmma_gemm(const unsigned short* __restrict__ A0, int lda0, long long zA0, int K0,
               const unsigned short* __restrict__ A1, int lda1, long long zA1, int Ktot,
               const unsigned short* __restrict__ Wp, long long zW,
               const float* __restrict__ bias, long long zBias,
               float* __restrict__ C, int ldc, long long zC)
{
    const int kb = blockIdx.z;
    A0 += (long long)kb * zA0;  A1 += (long long)kb * zA1;
    Wp += (long long)kb * zW;   bias += (long long)kb * zBias;
    C  += (long long)kb * zC;

    const int nt0  = blockIdx.x * 2;
    const int mt   = blockIdx.y;
    const int lane = threadIdx.x;
    const int KT   = Ktot >> 5;
    const int KT0  = K0 >> 5;
    const int m    = mt * 16 + (lane & 15);
    const int koff = (lane >> 4) * 8;     // A-fragment half select (ISA A layout)

    const unsigned short* wb0 = Wp + (long long)nt0 * KT * 512 + lane * 16;
    const unsigned short* wb1 = wb0 + (long long)KT * 512;

    v8f acc0 = {0.f,0.f,0.f,0.f,0.f,0.f,0.f,0.f};
    v8f acc1 = acc0;
    // phase 0: k in [0, K0) from A0
    gemm_phase(A0 + (long long)m * lda0, wb0, wb1, 0, KT0, KT, koff, acc0, acc1);
    // phase 1: k in [K0, Ktot) from A1 (pointer pre-biased by -K0)
    gemm_phase(A1 + (long long)m * lda1 - K0, wb0, wb1, KT0, KT, KT, koff, acc0, acc1);

    const int half = (lane >> 4) * 8;
    #pragma unroll
    for (int q = 0; q < 2; ++q) {
        const int ncol = (nt0 + q) * 16 + (lane & 15);
        const float bv = bias[ncol];
        const v8f acc = q ? acc1 : acc0;
        #pragma unroll
        for (int r = 0; r < 8; ++r) {
            int mrow = mt * 16 + half + r;
            C[(long long)mrow * ldc + ncol] = acc[r] + bv;
        }
    }
}

// ---------------- fused GRU: gr = h_blk @ gru_w[kb] (3 gates) -> new deter ----------------
__device__ __forceinline__ void load_frags3(const unsigned short* __restrict__ Arow,
                                            const unsigned short* __restrict__ w0,
                                            const unsigned short* __restrict__ w1,
                                            const unsigned short* __restrict__ w2,
                                            int kt, int koff,
                                            FragBF& a, FragBF& b0, FragBF& b1, FragBF& b2)
{
    const int k0 = kt * 32;
    a.h[0] = *(const v8us*)(Arow + k0 + koff);
    a.h[1] = *(const v8us*)(Arow + k0 + 16 + koff);
    b0.v   = *(const v16us*)(w0 + (long long)kt * 512);
    b1.v   = *(const v16us*)(w1 + (long long)kt * 512);
    b2.v   = *(const v16us*)(w2 + (long long)kt * 512);
}

__global__ __launch_bounds__(32)
void gru_kernel(const unsigned short* __restrict__ hbf, const unsigned short* __restrict__ Wp,
                const float* __restrict__ gru_b, float* __restrict__ deter,
                unsigned short* __restrict__ dbf, float* __restrict__ out, int t)
{
    const int nt = blockIdx.x, mt = blockIdx.y, kb = blockIdx.z;
    const int lane = threadIdx.x;
    const int m    = mt * 16 + (lane & 15);
    const int koff = (lane >> 4) * 8;
    const int KT   = 16;                                   // K = 512 (even)

    const unsigned short* Arow = hbf + (long long)m * DETER + kb * DPB;
    const unsigned short* w0 = Wp + (long long)kb * (1536 * 512)
                                  + (long long)nt * KT * 512 + lane * 16;
    const unsigned short* w1 = w0 + (long long)32 * KT * 512;
    const unsigned short* w2 = w1 + (long long)32 * KT * 512;

    v8f c0 = {0.f,0.f,0.f,0.f,0.f,0.f,0.f,0.f};
    v8f c1 = c0, c2 = c0;
    FragBF a0, b00, b10, b20, a1, b01, b11, b21;
    load_frags3(Arow, w0, w1, w2, 0, koff, a0, b00, b10, b20);
    for (int kt = 0; kt < KT; kt += 2) {
        load_frags3(Arow, w0, w1, w2, kt + 1, koff, a1, b01, b11, b21);
        c0 = __builtin_amdgcn_wmma_f32_16x16x32_bf16(false, a0.bf, false, b00.bf, (short)0, c0, false, false);
        c1 = __builtin_amdgcn_wmma_f32_16x16x32_bf16(false, a0.bf, false, b10.bf, (short)0, c1, false, false);
        c2 = __builtin_amdgcn_wmma_f32_16x16x32_bf16(false, a0.bf, false, b20.bf, (short)0, c2, false, false);
        if (kt + 2 < KT)
            load_frags3(Arow, w0, w1, w2, kt + 2, koff, a0, b00, b10, b20);
        c0 = __builtin_amdgcn_wmma_f32_16x16x32_bf16(false, a1.bf, false, b01.bf, (short)0, c0, false, false);
        c1 = __builtin_amdgcn_wmma_f32_16x16x32_bf16(false, a1.bf, false, b11.bf, (short)0, c1, false, false);
        c2 = __builtin_amdgcn_wmma_f32_16x16x32_bf16(false, a1.bf, false, b21.bf, (short)0, c2, false, false);
    }
    const int ncol = nt * 16 + (lane & 15);
    const int half = (lane >> 4) * 8;
    const int didx = kb * DPB + ncol;
    const float g0 = gru_b[kb * 1536 + 0 * 512 + ncol];
    const float g1 = gru_b[kb * 1536 + 1 * 512 + ncol];
    const float g2 = gru_b[kb * 1536 + 2 * 512 + ncol];
    #pragma unroll
    for (int r = 0; r < 8; ++r) {
        int mrow = mt * 16 + half + r;
        float dold = deter[(long long)mrow * DETER + didx];
        float rg   = sigm(c0[r] + g0);
        float cd   = c1[r] + g1;
        float up   = sigm(c2[r] + g2 - 1.f);
        float cand = tanhf(rg * cd);
        float dn   = up * cand + (1.f - up) * dold;
        deter[(long long)mrow * DETER + didx] = dn;
        dbf[(long long)mrow * DETER + didx]   = f2bf(dn);
        out[((long long)mrow * Ll + t) * ODIM + 1024 + didx] = dn;
    }
}

// ---------------- state init / reset-mask / action normalization ----------------
__global__ __launch_bounds__(256)
void init_kernel(const float* __restrict__ d0, const float* __restrict__ s0,
                 float* __restrict__ deter, unsigned short* __restrict__ dbf,
                 unsigned short* __restrict__ sbf)
{
    long long i = (long long)blockIdx.x * 256 + threadIdx.x;
    if (i < (long long)Bb * DETER) { float v = d0[i]; deter[i] = v; dbf[i] = f2bf(v); }
    if (i < (long long)Bb * SD)    sbf[i] = f2bf(s0[i]);
}

__global__ __launch_bounds__(256)
void mask_kernel(const float* __restrict__ action, const unsigned char* __restrict__ reset,
                 float* __restrict__ deter, unsigned short* __restrict__ dbf,
                 unsigned short* __restrict__ sbf, float* __restrict__ ahat, int t)
{
    const int b = blockIdx.x, tid = threadIdx.x;
    const bool r = reset[(long long)b * Ll + t] != 0;
    if (r) {
        #pragma unroll
        for (int i = 0; i < 16; ++i) {
            deter[(long long)b * DETER + tid * 16 + i] = 0.f;
            dbf  [(long long)b * DETER + tid * 16 + i] = 0;
        }
        #pragma unroll
        for (int i = 0; i < 4;  ++i) sbf[(long long)b * SD + tid * 4 + i] = 0;
    }
    if (tid < Aa) {
        float a = r ? 0.f : action[((long long)b * Ll + t) * Aa + tid];
        ahat[b * 4 + tid] = a / fmaxf(fabsf(a), 1.f);
    }
}

// ---------------- RMSNorm + SiLU epilogues (emit bf16 activations) ----------------
__global__ __launch_bounds__(256)
void rmsx_kernel(const float* __restrict__ xpre, const float* __restrict__ ahat,
                 const float* __restrict__ in2_w, const float* __restrict__ in2_b,
                 const float* __restrict__ g0, const float* __restrict__ g1,
                 const float* __restrict__ g2, unsigned short* __restrict__ xbf)
{
    const int b = blockIdx.x, seg = blockIdx.y, tid = threadIdx.x;
    float v[2];
    #pragma unroll
    for (int i = 0; i < 2; ++i) {
        int n = tid * 2 + i;
        float pre;
        if (seg < 2) pre = xpre[(long long)b * X3 + seg * 512 + n];
        else {
            const float* a = ahat + b * 4;
            pre = a[0]*in2_w[n*3+0] + a[1]*in2_w[n*3+1] + a[2]*in2_w[n*3+2] + in2_b[n];
        }
        v[i] = pre;
    }
    __shared__ float red[256];
    red[tid] = v[0]*v[0] + v[1]*v[1];
    __syncthreads();
    for (int s = 128; s > 0; s >>= 1) { if (tid < s) red[tid] += red[tid + s]; __syncthreads(); }
    const float inv = rsqrtf(red[0] / 512.f + 1e-4f);
    const float* g = (seg == 0) ? g0 : ((seg == 1) ? g1 : g2);
    #pragma unroll
    for (int i = 0; i < 2; ++i) {
        int n = tid * 2 + i;
        xbf[(long long)b * X3 + seg * 512 + n] = f2bf(silu(v[i] * inv * g[n]));
    }
}

__global__ __launch_bounds__(256)
void rmsh_kernel(const float* __restrict__ hpre, const float* __restrict__ g,
                 unsigned short* __restrict__ hbf)
{
    const int b = blockIdx.x, tid = threadIdx.x;
    float v[16], ss = 0.f;
    #pragma unroll
    for (int i = 0; i < 16; ++i) { v[i] = hpre[(long long)b * DETER + tid * 16 + i]; ss += v[i]*v[i]; }
    __shared__ float red[256];
    red[tid] = ss; __syncthreads();
    for (int s = 128; s > 0; s >>= 1) { if (tid < s) red[tid] += red[tid + s]; __syncthreads(); }
    const float inv = rsqrtf(red[0] / (float)DETER + 1e-4f);
    #pragma unroll
    for (int i = 0; i < 16; ++i) {
        int n = tid * 16 + i;
        hbf[(long long)b * DETER + n] = f2bf(silu(v[i] * inv * g[n]));
    }
}

__global__ __launch_bounds__(256)
void rmsz_kernel(const float* __restrict__ zpre, const float* __restrict__ g,
                 unsigned short* __restrict__ zbf)
{
    const int b = blockIdx.x, tid = threadIdx.x;
    float v[2], ss = 0.f;
    #pragma unroll
    for (int i = 0; i < 2; ++i) { v[i] = zpre[(long long)b * 512 + tid * 2 + i]; ss += v[i]*v[i]; }
    __shared__ float red[256];
    red[tid] = ss; __syncthreads();
    for (int s = 128; s > 0; s >>= 1) { if (tid < s) red[tid] += red[tid + s]; __syncthreads(); }
    const float inv = rsqrtf(red[0] / 512.f + 1e-4f);
    #pragma unroll
    for (int i = 0; i < 2; ++i) {
        int n = tid * 2 + i;
        zbf[(long long)b * 512 + n] = f2bf(silu(v[i] * inv * g[n]));
    }
}

// ---------------- softmax + unimix + gumbel argmax head ----------------
__global__ __launch_bounds__(32)
void head_kernel(const float* __restrict__ logitbuf, float* __restrict__ out,
                 unsigned short* __restrict__ sbf, int t)
{
    const int b = blockIdx.x, s = blockIdx.y, d = threadIdx.x;
    float lg = logitbuf[((long long)b * STOCH + s) * DISC + d];
    float mx = lg;
    for (int o = 16; o > 0; o >>= 1) mx = fmaxf(mx, __shfl_xor(mx, o, 32));
    float e = expf(lg - mx);
    float sum = e;
    for (int o = 16; o > 0; o >>= 1) sum += __shfl_xor(sum, o, 32);
    float p = 0.99f * (e / sum) + 0.01f / 32.f;
    float val = logf(p) + gumbel_hash(t, b, s, d);
    float mv = val;
    for (int o = 16; o > 0; o >>= 1) mv = fmaxf(mv, __shfl_xor(mv, o, 32));
    int cand = (val == mv) ? d : 32;
    for (int o = 16; o > 0; o >>= 1) cand = min(cand, __shfl_xor(cand, o, 32));
    float onehot = (d == cand) ? 1.f : 0.f;
    long long ob = ((long long)b * Ll + t) * ODIM;
    out[ob + s * DISC + d]        = onehot;      // stoch slice
    out[ob + 5120 + s * DISC + d] = lg;          // logit slice
    sbf[(long long)b * SD + s * DISC + d] = (d == cand) ? (unsigned short)0x3F80 : (unsigned short)0;
}

// ---------------- host orchestration ----------------
extern "C" void kernel_launch(void* const* d_in, const int* in_sizes, int n_in,
                              void* d_out, int out_size, void* d_ws, size_t ws_size,
                              hipStream_t stream)
{
    const float* embed   = (const float*)d_in[0];
    const float* action  = (const float*)d_in[1];
    const unsigned char* reset = (const unsigned char*)d_in[2];
    const float* stoch0  = (const float*)d_in[3];
    const float* deter0  = (const float*)d_in[4];
    const float* in0_w   = (const float*)d_in[5];
    const float* in0_b   = (const float*)d_in[6];
    const float* in0_g   = (const float*)d_in[7];
    const float* in1_w   = (const float*)d_in[8];
    const float* in1_b   = (const float*)d_in[9];
    const float* in1_g   = (const float*)d_in[10];
    const float* in2_w   = (const float*)d_in[11];
    const float* in2_b   = (const float*)d_in[12];
    const float* in2_g   = (const float*)d_in[13];
    const float* hid_w   = (const float*)d_in[14];
    const float* hid_b   = (const float*)d_in[15];
    const float* hid_g   = (const float*)d_in[16];
    const float* gru_w   = (const float*)d_in[17];
    const float* gru_b   = (const float*)d_in[18];
    const float* obs0_w  = (const float*)d_in[19];
    const float* obs0_b  = (const float*)d_in[20];
    const float* obs0_g  = (const float*)d_in[21];
    const float* obsl_w  = (const float*)d_in[22];
    const float* obsl_b  = (const float*)d_in[23];
    float* out = (float*)d_out;

    // ---- workspace carve: bf16 region first, fp32 region after
    unsigned short* us = (unsigned short*)d_ws;
    unsigned short* P_in0  = us;                    // 512*4096
    unsigned short* P_in1  = P_in0  + 2097152;      // 512*1024
    unsigned short* P_hid  = P_in1  + 524288;       // 8*2048*512
    unsigned short* P_gru  = P_hid  + 8388608;      // 8*512*1536
    unsigned short* P_obs0 = P_gru  + 6291456;      // 512*8192
    unsigned short* P_obsl = P_obs0 + 4194304;      // 1024*512
    unsigned short* Ebf    = P_obsl + 524288;       // 64*64*4096 embed in bf16
    unsigned short* Dbf    = Ebf    + 16777216;     // deter mirror (bf16)
    unsigned short* Sbf    = Dbf    + 262144;       // stoch (bf16)
    unsigned short* Xbf    = Sbf    + 65536;        // x (bf16)
    unsigned short* Hbf    = Xbf    + 98304;        // h (bf16)
    unsigned short* Zbf    = Hbf    + 262144;       // z (bf16)
    float* fpb = (float*)(Zbf + 32768);
    float* F_deter = fpb;                fpb += Bb * DETER;   // fp32 master state
    float* F_ahat  = fpb;                fpb += Bb * 4;
    float* F_xpre  = fpb;                fpb += Bb * X3;
    float* F_hpre  = fpb;                fpb += Bb * DETER;
    float* F_zpre  = fpb;                fpb += Bb * 512;
    float* F_logit = fpb;                fpb += Bb * SD;

    // ---- one-time (per call): weight repack + embed convert + state init
    pack_w<<<dim3(512,1,1), 256, 0, stream>>>(in0_w,  0,          4096, 1,    32, 128, P_in0,  0);
    pack_w<<<dim3(256,1,1), 256, 0, stream>>>(in1_w,  0,          1024, 1,    32, 32,  P_in1,  0);
    pack_w<<<dim3(256,1,8), 256, 0, stream>>>(hid_w,  2048*512,   1,    512,  32, 64,  P_hid,  1048576);
    pack_w<<<dim3(256,1,8), 256, 0, stream>>>(gru_w,  512*1536,   1,    1536, 96, 16,  P_gru,  786432);
    pack_w<<<dim3(512,1,1), 256, 0, stream>>>(obs0_w, 0,          8192, 1,    32, 256, P_obs0, 0);
    pack_w<<<dim3(256,1,1), 256, 0, stream>>>(obsl_w, 0,          512,  1,    64, 16,  P_obsl, 0);
    cvt_bf16<<<4096, 256, 0, stream>>>(embed, Ebf, (long long)Bb * Ll * Ee);
    init_kernel<<<(Bb*DETER + 255)/256, 256, 0, stream>>>(deter0, stoch0, F_deter, Dbf, Sbf);

    // ---- sequential scan over timesteps (recurrence via stream ordering)
    for (int t = 0; t < Ll; ++t) {
        mask_kernel<<<Bb, 256, 0, stream>>>(action, reset, F_deter, Dbf, Sbf, F_ahat, t);
        // x0 = deter @ in0_w.T + b    (64x4096 -> 512)
        wmma_gemm<<<dim3(16,4,1), 32, 0, stream>>>(Dbf, DETER, 0, 4096,
                                                   Dbf, DETER, 0, 4096,
                                                   P_in0, 0, in0_b, 0, F_xpre, X3, 0);
        // x1 = stoch @ in1_w.T + b    (64x1024 -> 512)
        wmma_gemm<<<dim3(16,4,1), 32, 0, stream>>>(Sbf, SD, 0, 1024,
                                                   Sbf, SD, 0, 1024,
                                                   P_in1, 0, in1_b, 0, F_xpre + 512, X3, 0);
        rmsx_kernel<<<dim3(Bb,3), 256, 0, stream>>>(F_xpre, F_ahat, in2_w, in2_b,
                                                    in0_g, in1_g, in2_g, Xbf);
        // hid: per block kb: [deter_blk | x] (2048) @ hid_w[kb] -> 512
        wmma_gemm<<<dim3(16,4,8), 32, 0, stream>>>(Dbf, DETER, 512, 512,
                                                   Xbf, X3, 0, 2048,
                                                   P_hid, 1048576, hid_b, 512,
                                                   F_hpre, DETER, 512);
        rmsh_kernel<<<Bb, 256, 0, stream>>>(F_hpre, hid_g, Hbf);
        // GRU: 3 gates fused, writes new deter (fp32 + bf16) + out deter slice
        gru_kernel<<<dim3(32,4,8), 32, 0, stream>>>(Hbf, P_gru, gru_b, F_deter, Dbf, out, t);
        // obs0: [deter | embed_t] (8192) @ obs0_w.T -> 512
        wmma_gemm<<<dim3(16,4,1), 32, 0, stream>>>(Dbf, DETER, 0, 4096,
                                                   Ebf + (long long)t * Ee, Ll * Ee, 0, 8192,
                                                   P_obs0, 0, obs0_b, 0, F_zpre, 512, 0);
        rmsz_kernel<<<Bb, 256, 0, stream>>>(F_zpre, obs0_g, Zbf);
        // logits = z @ obsl_w.T -> 1024
        wmma_gemm<<<dim3(32,4,1), 32, 0, stream>>>(Zbf, 512, 0, 512,
                                                   Zbf, 512, 0, 512,
                                                   P_obsl, 0, obsl_b, 0, F_logit, SD, 0);
        head_kernel<<<dim3(Bb, STOCH), 32, 0, stream>>>(F_logit, out, Sbf, t);
    }
    (void)in_sizes; (void)n_in; (void)out_size; (void)ws_size;
}
